// LayerRbf_82300163326748
// MI455X (gfx1250) — matrix operand order repaired and checked by reading, not compiled, not run on hardware
//
#include <hip/hip_runtime.h>

typedef __attribute__((ext_vector_type(16))) __bf16 v16bf;
typedef __attribute__((ext_vector_type(8)))  __bf16 v8bf;
typedef __attribute__((ext_vector_type(8)))  float  v8f;
typedef __attribute__((ext_vector_type(4)))  float  v4f;

#define B_ 8192
#define D_ 512
#define C_ 4096

// ---- workspace layout (bytes) ----
static constexpr size_t OFF_XHI  = 0;
static constexpr size_t OFF_XLO  = OFF_XHI  + (size_t)B_ * D_ * 2;
static constexpr size_t OFF_WTHI = OFF_XLO  + (size_t)B_ * D_ * 2;
static constexpr size_t OFF_WTLO = OFF_WTHI + (size_t)C_ * D_ * 2;
static constexpr size_t OFF_X2   = OFF_WTLO + (size_t)C_ * D_ * 2;
static constexpr size_t OFF_W2   = OFF_X2   + (size_t)B_ * 4;
static constexpr size_t OFF_RS   = OFF_W2   + (size_t)C_ * 4;

// ---------------------------------------------------------------------------
// Pass 1: split x (fp32) into bf16 hi + lo planes.
// ---------------------------------------------------------------------------
__global__ void rbf_cvt_x(const float* __restrict__ x,
                          __bf16* __restrict__ xhi, __bf16* __restrict__ xlo) {
  int i = (blockIdx.x * blockDim.x + threadIdx.x) * 4;
  v4f v = *(const v4f*)(x + i);
#pragma unroll
  for (int j = 0; j < 4; ++j) {
    __bf16 h = (__bf16)v[j];
    float  r = v[j] - (float)h;
    xhi[i + j] = h;
    xlo[i + j] = (__bf16)r;
  }
}

// ---------------------------------------------------------------------------
// Pass 2: transpose w[D][C] -> wT[C][D] while splitting into bf16 hi/lo.
// ---------------------------------------------------------------------------
__global__ void rbf_twc(const float* __restrict__ w,
                        __bf16* __restrict__ wthi, __bf16* __restrict__ wtlo) {
  __shared__ float tile[32][33];
  int c0 = blockIdx.x * 32, d0 = blockIdx.y * 32;
  int tx = threadIdx.x, ty = threadIdx.y;  // block (32, 8)
#pragma unroll
  for (int j = 0; j < 4; ++j)
    tile[ty + j * 8][tx] = w[(size_t)(d0 + ty + j * 8) * C_ + c0 + tx];
  __syncthreads();
#pragma unroll
  for (int j = 0; j < 4; ++j) {
    int c = ty + j * 8;
    float v = tile[tx][c];                  // = w[d0+tx][c0+c]
    __bf16 h = (__bf16)v;
    float  r = v - (float)h;
    size_t o = (size_t)(c0 + c) * D_ + d0 + tx;  // coalesced over tx
    wthi[o] = h;
    wtlo[o] = (__bf16)r;
  }
}

// ---------------------------------------------------------------------------
// Pass 3: w2[c] = sum_d w[d][c]^2 (exact fp32), rs[c] = 1/(2 s[c]^2).
// ---------------------------------------------------------------------------
__global__ void rbf_w2(const float* __restrict__ w, const float* __restrict__ s,
                       float* __restrict__ w2, float* __restrict__ rs) {
  int c = blockIdx.x * 256 + threadIdx.x;
  float acc = 0.f;
  for (int d = 0; d < D_; ++d) {
    float v = w[(size_t)d * C_ + c];
    acc += v * v;
  }
  w2[c] = acc;
  float sv = s[c];
  rs[c] = 0.5f / (sv * sv);
}

// ---------------------------------------------------------------------------
// Pass 4: x2[b] = sum_d x[b][d]^2 (exact fp32). One wave32 per row.
// ---------------------------------------------------------------------------
__global__ void rbf_x2(const float* __restrict__ x, float* __restrict__ x2) {
  int wid = threadIdx.x >> 5, lane = threadIdx.x & 31;
  int b = blockIdx.x * 8 + wid;
  const float* row = x + (size_t)b * D_;
  float acc = 0.f;
  for (int d = lane; d < D_; d += 32) {
    float v = row[d];
    acc += v * v;
  }
#pragma unroll
  for (int off = 16; off; off >>= 1) acc += __shfl_xor(acc, off, 32);
  if (lane == 0) x2[b] = acc;
}

// ---------------------------------------------------------------------------
// CDNA5 async global->LDS copy (ASYNCcnt-tracked, bypasses VGPRs).
// LDS operand = low 32 bits of the generic pointer.
// ---------------------------------------------------------------------------
__device__ __forceinline__ void async_copy_b128(const void* lds_ptr, const void* gptr) {
  unsigned l = (unsigned)(unsigned long long)lds_ptr;
  asm volatile("global_load_async_to_lds_b128 %0, %1, off"
               :: "v"(l), "v"(gptr)
               : "memory");
}
__device__ __forceinline__ void wait_async0() {
  asm volatile("s_wait_asynccnt 0x0" ::: "memory");
}

constexpr int LDSTR = 40;  // 32 K-elems + 8 pad; 80B stride: 16B aligned, conflict-free
struct Tile {
  __bf16 xh[128 * LDSTR];
  __bf16 xl[128 * LDSTR];
  __bf16 wh[128 * LDSTR];
  __bf16 wl[128 * LDSTR];
};

// ---------------------------------------------------------------------------
// Main: split-bf16 WMMA GEMM, double-buffered async-LDS pipeline, fused RBF
// epilogue. Block = 256 threads (8 waves, 2Mx4N), 128x128 tile; wave = 64x32.
// ---------------------------------------------------------------------------
__launch_bounds__(256)
__global__ void rbf_main(const __bf16* __restrict__ xhi, const __bf16* __restrict__ xlo,
                         const __bf16* __restrict__ wthi, const __bf16* __restrict__ wtlo,
                         const float* __restrict__ x2, const float* __restrict__ w2,
                         const float* __restrict__ rs, float* __restrict__ out) {
  __shared__ __align__(16) Tile tiles[2];  // 2 x 40KB of 320KB/WGP

  const int t = threadIdx.x;
  const int lane = t & 31, wid = t >> 5;
  const int waveM = wid & 1, waveN = wid >> 1;
  const int blockN = blockIdx.x * 128;
  const int blockM = blockIdx.y * 128;

  v8f acc[4][2] = {};

  const int half = lane >> 4;      // which 16-lane half
  const int lrow = lane & 15;
  const int ka = half * 8;         // A-frag chunk base K (ISA 16-bit A layout)
  const int kb = half * 8;         // B-frag chunk base K (same layout as A)

  // each thread owns 2 of the 512 16B-chunks per plane
  const int idx0 = t, idx1 = t + 256;
  const int row0 = idx0 >> 2, ch0 = (idx0 & 3) * 8;
  const int row1 = idx1 >> 2, ch1 = (idx1 & 3) * 8;

  auto fill = [&](Tile* tl, int k0) {
    size_t gx0 = (size_t)(blockM + row0) * D_ + k0 + ch0;
    size_t gw0 = (size_t)(blockN + row0) * D_ + k0 + ch0;
    size_t gx1 = (size_t)(blockM + row1) * D_ + k0 + ch1;
    size_t gw1 = (size_t)(blockN + row1) * D_ + k0 + ch1;
    int l0 = row0 * LDSTR + ch0, l1 = row1 * LDSTR + ch1;
    async_copy_b128(&tl->xh[l0], xhi + gx0);
    async_copy_b128(&tl->xl[l0], xlo + gx0);
    async_copy_b128(&tl->wh[l0], wthi + gw0);
    async_copy_b128(&tl->wl[l0], wtlo + gw0);
    async_copy_b128(&tl->xh[l1], xhi + gx1);
    async_copy_b128(&tl->xl[l1], xlo + gx1);
    async_copy_b128(&tl->wh[l1], wthi + gw1);
    async_copy_b128(&tl->wl[l1], wtlo + gw1);
  };

  fill(&tiles[0], 0);  // prologue

  constexpr int NIT = D_ / 32;
  for (int it = 0; it < NIT; ++it) {
    Tile* cur = &tiles[it & 1];
    wait_async0();        // my 8 async copies into cur are done
    __syncthreads();      // everyone's copies done; everyone done reading other buf
    if (it + 1 < NIT) fill(&tiles[(it + 1) & 1], (it + 1) * 32);  // overlap w/ compute

    v16bf ah[4], al[4], bh[2], bl[2];
    union U { v16bf v; v8bf h[2]; };
#pragma unroll
    for (int mi = 0; mi < 4; ++mi) {
      int r = waveM * 64 + mi * 16 + lrow;
      U u;
      u.h[0] = *(const v8bf*)&cur->xh[r * LDSTR + ka];
      u.h[1] = *(const v8bf*)&cur->xh[r * LDSTR + ka + 16];
      ah[mi] = u.v;
      u.h[0] = *(const v8bf*)&cur->xl[r * LDSTR + ka];
      u.h[1] = *(const v8bf*)&cur->xl[r * LDSTR + ka + 16];
      al[mi] = u.v;
    }
#pragma unroll
    for (int ni = 0; ni < 2; ++ni) {
      int r = waveN * 32 + ni * 16 + lrow;
      U u;
      u.h[0] = *(const v8bf*)&cur->wh[r * LDSTR + kb];
      u.h[1] = *(const v8bf*)&cur->wh[r * LDSTR + kb + 16];
      bh[ni] = u.v;
      u.h[0] = *(const v8bf*)&cur->wl[r * LDSTR + kb];
      u.h[1] = *(const v8bf*)&cur->wl[r * LDSTR + kb + 16];
      bl[ni] = u.v;
    }

#pragma unroll
    for (int mi = 0; mi < 4; ++mi)
#pragma unroll
      for (int ni = 0; ni < 2; ++ni) {
        // x*w ~= hi*hi + hi*lo + lo*hi  (lo*lo ~ 2^-16 relative, dropped)
        acc[mi][ni] = __builtin_amdgcn_wmma_f32_16x16x32_bf16(
            false, ah[mi], false, bh[ni], (short)0, acc[mi][ni], false, false);
        acc[mi][ni] = __builtin_amdgcn_wmma_f32_16x16x32_bf16(
            false, ah[mi], false, bl[ni], (short)0, acc[mi][ni], false, false);
        acc[mi][ni] = __builtin_amdgcn_wmma_f32_16x16x32_bf16(
            false, al[mi], false, bh[ni], (short)0, acc[mi][ni], false, false);
      }
  }

  // Fused epilogue: dist = x2 + w2 - 2*xw ; out = exp(-dist * 1/(2 s^2))
#pragma unroll
  for (int mi = 0; mi < 4; ++mi) {
#pragma unroll
    for (int ni = 0; ni < 2; ++ni) {
      int col = blockN + waveN * 32 + ni * 16 + (lane & 15);
      float w2c = w2[col], rsc = rs[col];
#pragma unroll
      for (int r = 0; r < 8; ++r) {
        int rowG = blockM + waveM * 64 + mi * 16 + r + (lane >> 4) * 8;
        float a = acc[mi][ni][r];
        float dist = x2[rowG] + w2c - 2.0f * a;
        out[(size_t)rowG * C_ + col] = __expf(-dist * rsc);
      }
    }
  }
}

// ---------------------------------------------------------------------------
extern "C" void kernel_launch(void* const* d_in, const int* in_sizes, int n_in,
                              void* d_out, int out_size, void* d_ws, size_t ws_size,
                              hipStream_t stream) {
  const float* x = (const float*)d_in[0];  // [B, D]
  const float* w = (const float*)d_in[1];  // [D, C]
  const float* s = (const float*)d_in[2];  // [1, C]
  float* out = (float*)d_out;              // [B, C]

  char* ws = (char*)d_ws;
  __bf16* xhi  = (__bf16*)(ws + OFF_XHI);
  __bf16* xlo  = (__bf16*)(ws + OFF_XLO);
  __bf16* wthi = (__bf16*)(ws + OFF_WTHI);
  __bf16* wtlo = (__bf16*)(ws + OFF_WTLO);
  float*  x2   = (float*)(ws + OFF_X2);
  float*  w2   = (float*)(ws + OFF_W2);
  float*  rs   = (float*)(ws + OFF_RS);

  rbf_cvt_x<<<(B_ * D_ / 4) / 256, 256, 0, stream>>>(x, xhi, xlo);
  rbf_twc<<<dim3(C_ / 32, D_ / 32), dim3(32, 8), 0, stream>>>(w, wthi, wtlo);
  rbf_w2<<<C_ / 256, 256, 0, stream>>>(w, s, w2, rs);
  rbf_x2<<<B_ / 8, 256, 0, stream>>>(x, x2);
  rbf_main<<<dim3(C_ / 128, B_ / 128), 256, 0, stream>>>(
      xhi, xlo, wthi, wtlo, x2, w2, rs, out);
}
